// GraphRegressionLayer_91250875170843
// MI455X (gfx1250) — compile-verified, matrix operand order checked by reading, not run on hardware
//
#include <hip/hip_runtime.h>
#include <hip/hip_bf16.h>
#include <math.h>

typedef __attribute__((ext_vector_type(16))) __bf16 v16bf;
typedef __attribute__((ext_vector_type(8)))  float  v8f;

#define ALPHA_C 0.1f
#define EPS_C   1e-5f

// ---------------------------------------------------------------- WMMA helpers
__device__ __forceinline__ v8f wmma_bf16(v16bf a, v16bf b, v8f c) {
  // D = A(16x32 bf16) * B(32x16 bf16) + C(16x16 f32)
  return __builtin_amdgcn_wmma_f32_16x16x32_bf16(false, a, false, b, (short)0, c, false, false);
}

// Row-major fragment load matching the CDNA5 16-bit A layout (wave32):
// lanes 0-15: row M=lane, K = {0..7, 16..23}; lanes 16-31: row M=lane-16, K = {8..15, 24..31}.
// Used for A, and for B given as W[n][k] row-major (B = W^T in K x N form).
__device__ __forceinline__ v16bf load_frag(const __bf16* __restrict__ base, int stride, int lane) {
  const int r  = lane & 15;
  const int kh = (lane >> 4) << 3;
  const __bf16* p = base + (size_t)r * stride;
  union { v16bf v; __bf16 h[16]; } u;
#pragma unroll
  for (int e = 0; e < 16; ++e) {
    const int k = ((e < 8) ? e : (e + 8)) + kh;
    u.h[e] = p[k];
  }
  return u.v;
}

// ---------------------------------------------------------------- prep kernels
__global__ void cvt_bf16_kernel(const float* __restrict__ in, __bf16* __restrict__ out, long n) {
  long i = (long)blockIdx.x * blockDim.x + threadIdx.x;
  if (i < n) out[i] = (__bf16)in[i];
}

// OIHW (64, CI, 3, 3) f32 -> [O][tap*CI + ci] bf16 (implicit-GEMM B layout)
__global__ void prep_conv_w_kernel(const float* __restrict__ w, __bf16* __restrict__ out, int CI) {
  int idx = blockIdx.x * blockDim.x + threadIdx.x;
  int total = 64 * CI * 9;
  if (idx >= total) return;
  int o  = idx / (CI * 9);
  int r  = idx % (CI * 9);
  int ci = r / 9;
  int t  = r % 9;
  out[(size_t)o * (9 * CI) + t * CI + ci] = (__bf16)w[idx];
}

__global__ void fill_u32_kernel(unsigned int* __restrict__ p, unsigned int v, long n) {
  long i = (long)blockIdx.x * blockDim.x + threadIdx.x;
  if (i < n) p[i] = v;
}

// ---------------------------------------------------------------- conv 3x3 SAME (implicit GEMM, bf16 WMMA)
// Channels-last input [Npix, CI_], output [Npix, 64] bf16, fused BN+ReLU.
// Each wave: 16 consecutive pixels (same image row) x 64 out channels, K = 9*CI_.
template <int CI_>
__global__ __launch_bounds__(256) void conv3x3_bn_relu_kernel(
    const __bf16* __restrict__ In, const __bf16* __restrict__ Wr,
    const float* __restrict__ g, const float* __restrict__ bt,
    const float* __restrict__ mn, const float* __restrict__ vr,
    __bf16* __restrict__ OutB, int Npix)
{
  constexpr int K9 = 9 * CI_;
  extern __shared__ __bf16 sW[];                       // [64][K9]
  for (int i = threadIdx.x; i < 64 * K9; i += 256) sW[i] = Wr[i];
  __syncthreads();

  const int lane    = threadIdx.x & 31;
  const int wave    = threadIdx.x >> 5;
  const int pixbase = blockIdx.x * 128 + wave * 16;
  if (pixbase >= Npix) return;

  const int n   = lane & 15;
  const int kh8 = (lane >> 4) << 3;
  const int px  = pixbase + n;                         // A-matrix row = pixel
  const int x   = px & 255;
  const int ygl = px >> 8;                             // global row (includes batch)
  const int y   = ygl & 255;

  __builtin_prefetch(In + (size_t)pixbase * CI_, 0, 0);

  v8f acc[4] = {v8f{}, v8f{}, v8f{}, v8f{}};

#pragma unroll
  for (int ky = 0; ky < 3; ++ky) {
    const int  sy  = y + ky - 1;
    const bool yok = (sy >= 0) && (sy < 256);
#pragma unroll
    for (int kx = 0; kx < 3; ++kx) {
      const int  sx = x + kx - 1;
      const bool ok = yok && (sx >= 0) && (sx < 256);
      const __bf16* src = In + ((size_t)(ygl + ky - 1) * 256 + sx) * CI_;
      const int tap = ky * 3 + kx;
#pragma unroll
      for (int ks = 0; ks < CI_ / 32; ++ks) {
        union { v16bf v; __bf16 h[16]; } ua;
#pragma unroll
        for (int e = 0; e < 16; ++e) {
          const int k = ks * 32 + ((e < 8) ? e : (e + 8)) + kh8;
          ua.h[e] = ok ? src[k] : (__bf16)0.0f;        // zero padding at borders
        }
#pragma unroll
        for (int nt = 0; nt < 4; ++nt) {
          v16bf bf = load_frag(sW + (size_t)(nt * 16) * K9 + tap * CI_ + ks * 32, K9, lane);
          acc[nt] = wmma_bf16(ua.v, bf, acc[nt]);
        }
      }
    }
  }

  // Epilogue: BN + ReLU, store bf16 (D layout: lane -> col n, VGPR e -> row e+kh8)
#pragma unroll
  for (int nt = 0; nt < 4; ++nt) {
    const int   c  = nt * 16 + n;
    const float sc = g[c] * rsqrtf(vr[c] + EPS_C);
    const float mu = mn[c], be = bt[c];
#pragma unroll
    for (int e = 0; e < 8; ++e) {
      float v2 = (acc[nt][e] - mu) * sc + be;
      v2 = fmaxf(v2, 0.0f);
      OutB[(size_t)(pixbase + kh8 + e) * 64 + c] = (__bf16)v2;
    }
  }
}

// ---------------------------------------------------------------- dense [rows,64] x [64,64]^T GEMM (bf16 WMMA)
__global__ __launch_bounds__(256) void gemm_n64_kernel(
    const __bf16* __restrict__ A, const __bf16* __restrict__ W,  // W: [out][in] row-major
    const float* __restrict__ bias, float* __restrict__ OutF,
    __bf16* __restrict__ OutB, int rows)
{
  extern __shared__ __bf16 sW[];                       // [64][64]
  for (int i = threadIdx.x; i < 64 * 64; i += 256) sW[i] = W[i];
  __syncthreads();

  const int lane    = threadIdx.x & 31;
  const int wave    = threadIdx.x >> 5;
  const int rowbase = blockIdx.x * 128 + wave * 16;
  if (rowbase >= rows) return;

  const __bf16* Ap = A + (size_t)rowbase * 64;
  const v16bf a0 = load_frag(Ap,      64, lane);       // K 0..31
  const v16bf a1 = load_frag(Ap + 32, 64, lane);       // K 32..63

  v8f acc[4] = {v8f{}, v8f{}, v8f{}, v8f{}};
#pragma unroll
  for (int nt = 0; nt < 4; ++nt) {
    v16bf b0 = load_frag(sW + (size_t)(nt * 16) * 64,      64, lane);
    v16bf b1 = load_frag(sW + (size_t)(nt * 16) * 64 + 32, 64, lane);
    acc[nt] = wmma_bf16(a0, b0, acc[nt]);
    acc[nt] = wmma_bf16(a1, b1, acc[nt]);
  }

  const int n  = lane & 15;
  const int mh = (lane >> 4) << 3;
#pragma unroll
  for (int nt = 0; nt < 4; ++nt) {
    const int   c = nt * 16 + n;
    const float b = bias ? bias[c] : 0.0f;
#pragma unroll
    for (int e = 0; e < 8; ++e) {
      const float  vv  = acc[nt][e] + b;
      const size_t off = (size_t)(rowbase + mh + e) * 64 + c;
      if (OutF) OutF[off] = vv;
      if (OutB) OutB[off] = (__bf16)vv;
    }
  }
}

// ---------------------------------------------------------------- graph kernels (memory/atomic bound, f32)
__global__ void deg_kernel(const int* __restrict__ col, const float* __restrict__ ea,
                           float* __restrict__ deg, int E, int Nn) {
  int i = blockIdx.x * blockDim.x + threadIdx.x;
  if (i < E) {
    atomicAdd(&deg[col[i]], ea[2 * (size_t)i + 1]);
  } else if (i < E + Nn) {
    atomicAdd(&deg[i - E], 2.0f);                      // self loop weight
  }
}

__global__ void dinv_kernel(float* __restrict__ deg, int Nn) {
  int i = blockIdx.x * blockDim.x + threadIdx.x;
  if (i >= Nn) return;
  float d = deg[i];
  deg[i] = (d > 0.0f) ? rsqrtf(d) : 0.0f;
}

__global__ void gcn_scatter_kernel(const int* __restrict__ row, const int* __restrict__ col,
                                   const float* __restrict__ ea, const float* __restrict__ dinv,
                                   const float* __restrict__ hl, float* __restrict__ hagg,
                                   int E, int Nn) {
  long idx = (long)blockIdx.x * blockDim.x + threadIdx.x;
  long total = (long)(E + Nn) * 64;
  if (idx >= total) return;
  int e  = (int)(idx >> 6);
  int ch = (int)(idx & 63);
  int r, c; float w;
  if (e < E) { r = row[e]; c = col[e]; w = ea[2 * (size_t)e + 1]; }
  else       { r = c = e - E; w = 2.0f; }
  float norm = dinv[r] * w * dinv[c];
  atomicAdd(&hagg[(size_t)c * 64 + ch], norm * hl[(size_t)r * 64 + ch]);
}

__global__ void bias_bn_elu_kernel(float* __restrict__ h, const float* __restrict__ bias,
                                   const float* __restrict__ g, const float* __restrict__ b,
                                   const float* __restrict__ m, const float* __restrict__ v,
                                   __bf16* __restrict__ outB, long total) {
  long i = (long)blockIdx.x * blockDim.x + threadIdx.x;
  if (i >= total) return;
  int c = (int)(i & 63);
  float x = h[i] + (bias ? bias[c] : 0.0f);
  x = (x - m[c]) * (g[c] * rsqrtf(v[c] + EPS_C)) + b[c];
  x = (x > 0.0f) ? x : ALPHA_C * expm1f(x);
  h[i] = x;
  if (outB) outB[i] = (__bf16)x;
}

__device__ __forceinline__ void atomicMaxF(float* addr, float val) {
  if (val >= 0.0f) atomicMax((int*)addr, __float_as_int(val));
  else             atomicMin((unsigned int*)addr, (unsigned int)__float_as_int(val));
}

__global__ void attn_logits_kernel(const int* __restrict__ row, const int* __restrict__ col,
                                   const float* __restrict__ ea, const float* __restrict__ q,
                                   const float* __restrict__ k, const float* __restrict__ tew,
                                   float* __restrict__ logits, float* __restrict__ mx, int E) {
  int e = blockIdx.x * blockDim.x + threadIdx.x;
  if (e >= E) return;
  int r = row[e], c = col[e];
  float a0 = ea[2 * (size_t)e + 0], a1 = ea[2 * (size_t)e + 1];
  const float* qp = q + (size_t)c * 64;
  const float* kp = k + (size_t)r * 64;
  float dot = 0.0f;
#pragma unroll 8
  for (int ch = 0; ch < 64; ++ch) {
    float kj = kp[ch] + a0 * tew[2 * ch] + a1 * tew[2 * ch + 1];
    dot += qp[ch] * kj;
  }
  float l = dot * 0.125f;                              // 1/sqrt(64)
  logits[e] = l;
  atomicMaxF(&mx[c], l);
}

__global__ void mx_fix_kernel(float* __restrict__ mx, int Nn) {
  int i = blockIdx.x * blockDim.x + threadIdx.x;
  if (i >= Nn) return;
  float m = mx[i];
  if (isinf(m) || isnan(m)) mx[i] = 0.0f;
}

__global__ void attn_exp_kernel(const int* __restrict__ col, float* __restrict__ logits,
                                const float* __restrict__ mx, float* __restrict__ ssum, int E) {
  int e = blockIdx.x * blockDim.x + threadIdx.x;
  if (e >= E) return;
  int c = col[e];
  float a = expf(logits[e] - mx[c]);
  logits[e] = a;
  atomicAdd(&ssum[c], a);
}

__global__ void attn_agg_kernel(const int* __restrict__ row, const int* __restrict__ col,
                                const float* __restrict__ ea, const float* __restrict__ aexp,
                                const float* __restrict__ ssum, const float* __restrict__ v,
                                const float* __restrict__ tew, float* __restrict__ agg, int E) {
  long idx = (long)blockIdx.x * blockDim.x + threadIdx.x;
  long total = (long)E * 64;
  if (idx >= total) return;
  int e  = (int)(idx >> 6);
  int ch = (int)(idx & 63);
  int r = row[e], c = col[e];
  float a  = aexp[e] / fmaxf(ssum[c], 1e-16f);
  float a0 = ea[2 * (size_t)e + 0], a1 = ea[2 * (size_t)e + 1];
  float vj = v[(size_t)r * 64 + ch] + a0 * tew[2 * ch] + a1 * tew[2 * ch + 1];
  atomicAdd(&agg[(size_t)c * 64 + ch], a * vj);
}

__global__ void final_kernel(const float* __restrict__ agg,
                             const float* __restrict__ g, const float* __restrict__ b,
                             const float* __restrict__ m, const float* __restrict__ v,
                             const float* __restrict__ linw, const float* __restrict__ linb,
                             float* __restrict__ out, int Nn) {
  int i = blockIdx.x * blockDim.x + threadIdx.x;
  if (i >= Nn) return;
  const float* hp = agg + (size_t)i * 64;
  float d0 = 0.0f, d1 = 0.0f;
#pragma unroll 8
  for (int ch = 0; ch < 64; ++ch) {
    float x = hp[ch];
    x = (x - m[ch]) * (g[ch] * rsqrtf(v[ch] + EPS_C)) + b[ch];
    x = (x > 0.0f) ? x : ALPHA_C * expm1f(x);
    d0 += x * linw[ch];
    d1 += x * linw[64 + ch];
  }
  out[2 * (size_t)i + 0] = d0 + linb[0];
  out[2 * (size_t)i + 1] = d1 + linb[1];
}

// ---------------------------------------------------------------- host launch
extern "C" void kernel_launch(void* const* d_in, const int* in_sizes, int n_in,
                              void* d_out, int out_size, void* d_ws, size_t ws_size,
                              hipStream_t stream) {
  const float* x    = (const float*)d_in[0];
  const int*   ei   = (const int*)d_in[1];
  const float* ea   = (const float*)d_in[2];
  const float* c1w  = (const float*)d_in[6];
  const float* c2w  = (const float*)d_in[7];
  const float* bg2  = (const float*)d_in[8];
  const float* bb2  = (const float*)d_in[9];
  const float* bm2  = (const float*)d_in[10];
  const float* bv2  = (const float*)d_in[11];
  const float* gcnw = (const float*)d_in[12];
  const float* gcnb = (const float*)d_in[13];
  const float* bgg  = (const float*)d_in[14];
  const float* bgb  = (const float*)d_in[15];
  const float* bgm  = (const float*)d_in[16];
  const float* bgv  = (const float*)d_in[17];
  const float* tqw  = (const float*)d_in[18];
  const float* tqb  = (const float*)d_in[19];
  const float* tkw  = (const float*)d_in[20];
  const float* tkb  = (const float*)d_in[21];
  const float* tvw  = (const float*)d_in[22];
  const float* tvb  = (const float*)d_in[23];
  const float* tew  = (const float*)d_in[24];
  const float* skw  = (const float*)d_in[25];
  const float* skb  = (const float*)d_in[26];
  const float* linw = (const float*)d_in[27];
  const float* linb = (const float*)d_in[28];

  const int Nn = in_sizes[0] / 32;                     // 262144
  const int E  = in_sizes[1] / 2;                      // 1048576
  const int* row = ei;
  const int* col = ei + E;
  float* out = (float*)d_out;

  // workspace carve-out
  char* w = (char*)d_ws;
  auto carve = [&](size_t bytes) {
    char* p = w;
    w += (bytes + 255) & ~(size_t)255;
    return p;
  };
  __bf16* xb    = (__bf16*)carve((size_t)Nn * 32 * 2);   // x in bf16
  __bf16* wc1   = (__bf16*)carve(64 * 288 * 2);          // conv1 weights [64][9*32]
  __bf16* wc2   = (__bf16*)carve(64 * 576 * 2);          // conv2 weights [64][9*64]
  __bf16* wgcn  = (__bf16*)carve(64 * 64 * 2);
  __bf16* wq    = (__bf16*)carve(64 * 64 * 2);
  __bf16* wk    = (__bf16*)carve(64 * 64 * 2);
  __bf16* wv    = (__bf16*)carve(64 * 64 * 2);
  __bf16* wsk   = (__bf16*)carve(64 * 64 * 2);
  __bf16* c1b   = (__bf16*)carve((size_t)Nn * 64 * 2);   // conv1 out
  __bf16* hb    = (__bf16*)carve((size_t)Nn * 64 * 2);   // conv2 out (graph features, bf16)
  float*  deg   = (float*)carve((size_t)Nn * 4);         // degree -> dinv in place
  float*  hl    = (float*)carve((size_t)Nn * 64 * 4);    // h @ gcn_w^T
  float*  hg    = (float*)carve((size_t)Nn * 64 * 4);    // gcn aggregate -> post-BN/ELU h
  __bf16* hgb   = (__bf16*)carve((size_t)Nn * 64 * 2);   // bf16 copy for q/k/v/skip GEMMs
  float*  qf    = (float*)carve((size_t)Nn * 64 * 4);
  float*  kf    = (float*)carve((size_t)Nn * 64 * 4);
  float*  vf    = (float*)carve((size_t)Nn * 64 * 4);
  float*  agg   = (float*)carve((size_t)Nn * 64 * 4);    // skip GEMM result + attention agg
  float*  aexp  = (float*)carve((size_t)E * 4);
  float*  mx    = (float*)carve((size_t)Nn * 4);
  float*  ssum  = (float*)carve((size_t)Nn * 4);
  (void)ws_size; (void)n_in; (void)out_size;

  const int TB = 256;
  auto blocks = [](long n, int tb) { return (unsigned int)((n + tb - 1) / tb); };

  // --- prep: bf16 conversions + conv weight relayout
  cvt_bf16_kernel<<<blocks((long)Nn * 32, TB), TB, 0, stream>>>(x, xb, (long)Nn * 32);
  prep_conv_w_kernel<<<blocks(64 * 32 * 9, TB), TB, 0, stream>>>(c1w, wc1, 32);
  prep_conv_w_kernel<<<blocks(64 * 64 * 9, TB), TB, 0, stream>>>(c2w, wc2, 64);
  cvt_bf16_kernel<<<blocks(4096, TB), TB, 0, stream>>>(gcnw, wgcn, 4096);
  cvt_bf16_kernel<<<blocks(4096, TB), TB, 0, stream>>>(tqw, wq, 4096);
  cvt_bf16_kernel<<<blocks(4096, TB), TB, 0, stream>>>(tkw, wk, 4096);
  cvt_bf16_kernel<<<blocks(4096, TB), TB, 0, stream>>>(tvw, wv, 4096);
  cvt_bf16_kernel<<<blocks(4096, TB), TB, 0, stream>>>(skw, wsk, 4096);

  // --- conv1 + BN + ReLU, conv2 + BN + ReLU (WMMA implicit GEMM)
  const unsigned int cblocks = (unsigned int)((Nn + 127) / 128);
  conv3x3_bn_relu_kernel<32><<<cblocks, TB, 64 * 288 * 2, stream>>>(xb, wc1, bg2, bb2, bm2, bv2, c1b, Nn);
  conv3x3_bn_relu_kernel<64><<<cblocks, TB, 64 * 576 * 2, stream>>>(c1b, wc2, bg2, bb2, bm2, bv2, hb, Nn);

  // --- GCN normalization: deg -> dinv
  fill_u32_kernel<<<blocks(Nn, TB), TB, 0, stream>>>((unsigned int*)deg, 0u, Nn);
  deg_kernel<<<blocks((long)E + Nn, TB), TB, 0, stream>>>(col, ea, deg, E, Nn);
  dinv_kernel<<<blocks(Nn, TB), TB, 0, stream>>>(deg, Nn);

  // --- hl = h @ gcn_w^T (WMMA), then normalized scatter
  gemm_n64_kernel<<<cblocks, TB, 64 * 64 * 2, stream>>>(hb, wgcn, nullptr, hl, nullptr, Nn);
  fill_u32_kernel<<<blocks((long)Nn * 64, TB), TB, 0, stream>>>((unsigned int*)hg, 0u, (long)Nn * 64);
  gcn_scatter_kernel<<<blocks((long)(E + Nn) * 64, TB), TB, 0, stream>>>(row, col, ea, deg, hl, hg, E, Nn);
  bias_bn_elu_kernel<<<blocks((long)Nn * 64, TB), TB, 0, stream>>>(hg, gcnb, bgg, bgb, bgm, bgv, hgb, (long)Nn * 64);

  // --- q/k/v/skip projections (WMMA); skip result seeds the attention accumulator
  gemm_n64_kernel<<<cblocks, TB, 64 * 64 * 2, stream>>>(hgb, wq, tqb, qf, nullptr, Nn);
  gemm_n64_kernel<<<cblocks, TB, 64 * 64 * 2, stream>>>(hgb, wk, tkb, kf, nullptr, Nn);
  gemm_n64_kernel<<<cblocks, TB, 64 * 64 * 2, stream>>>(hgb, wv, tvb, vf, nullptr, Nn);
  gemm_n64_kernel<<<cblocks, TB, 64 * 64 * 2, stream>>>(hgb, wsk, skb, agg, nullptr, Nn);

  // --- edge softmax attention
  fill_u32_kernel<<<blocks(Nn, TB), TB, 0, stream>>>((unsigned int*)mx, 0xFF800000u, Nn);   // -inf
  fill_u32_kernel<<<blocks(Nn, TB), TB, 0, stream>>>((unsigned int*)ssum, 0u, Nn);
  attn_logits_kernel<<<blocks(E, TB), TB, 0, stream>>>(row, col, ea, qf, kf, tew, aexp, mx, E);
  mx_fix_kernel<<<blocks(Nn, TB), TB, 0, stream>>>(mx, Nn);
  attn_exp_kernel<<<blocks(E, TB), TB, 0, stream>>>(col, aexp, mx, ssum, E);
  attn_agg_kernel<<<blocks((long)E * 64, TB), TB, 0, stream>>>(row, col, ea, aexp, ssum, vf, tew, agg, E);

  // --- final BN + ELU + 64->2 linear
  final_kernel<<<blocks(Nn, TB), TB, 0, stream>>>(agg, bgg, bgb, bgm, bgv, linw, linb, out, Nn);
}
